// aggregator_16922171146281
// MI455X (gfx1250) — matrix operand order repaired and checked by reading, not compiled, not run on hardware
//
#include <hip/hip_runtime.h>
#include <hip/hip_bf16.h>

#define FEAT 128
#define EMB 64
#define KNEI 24
#define LDS_STRIDE 130   // 128 + 2 halfword pad -> bank-conflict-free strided reads

typedef __attribute__((ext_vector_type(16))) __bf16 v16bf;
typedef __attribute__((ext_vector_type(8)))  float  v8f;

union Frag16 { uint32_t w[8]; v16bf v; };

// -------------------------------------------------------------------------
// Zero-fill kernel (embed_matrix half of d_out must start at 0 every call)
// -------------------------------------------------------------------------
__global__ __launch_bounds__(256) void zero_f4_kernel(float4* __restrict__ p, int n4) {
    int i = blockIdx.x * blockDim.x + threadIdx.x;
    if (i < n4) p[i] = make_float4(0.f, 0.f, 0.f, 0.f);
}

// -------------------------------------------------------------------------
// Gathered GEMM:  out[r, :] = A[idx ? idx[r] : r, :] @ W + bias
// A: [*, 128] f32, W: [128, 64] f32, out: [nrows, 64] f32
// 4 waves / block, one 16x64 output tile per wave.
// Per wave: 4 K-steps x 4 N-tiles of v_wmma_f32_16x16x32_bf16 (f32 accum).
// -------------------------------------------------------------------------
__global__ __launch_bounds__(128) void proj_gemm_kernel(
    const float* __restrict__ A, const int* __restrict__ idx,
    const float* __restrict__ W, const float* __restrict__ bias,
    float* __restrict__ out, int nrows)
{
    __shared__ __align__(16) __bf16 sA[4 * 16 * LDS_STRIDE];  // 4 wave tiles, 16x128
    __shared__ __align__(16) __bf16 sBt[EMB * LDS_STRIDE];    // W transposed: [64][128]

    const int tid   = threadIdx.x;
    const int wslot = tid >> 5;
    const int lane  = tid & 31;

    // Stage W^T (bf16) once per block: sBt[n][k] = W[k][n]
    for (int e = tid; e < FEAT * EMB; e += 128) {
        int k = e >> 6;   // 0..127
        int n = e & 63;   // 0..63
        sBt[n * LDS_STRIDE + k] = (__bf16)W[e];
    }
    // Stage 16 (gathered) A rows per wave as bf16
    const int tile0 = (blockIdx.x * 4 + wslot) * 16;
    for (int e = lane; e < 16 * FEAT; e += 32) {
        int r = e >> 7;
        int c = e & 127;                      // consecutive lanes -> coalesced
        int grow = tile0 + r;
        if (grow > nrows - 1) grow = nrows - 1;
        int src = idx ? idx[grow] : grow;
        sA[wslot * (16 * LDS_STRIDE) + r * LDS_STRIDE + c] = (__bf16)A[(size_t)src * FEAT + c];
    }
    __syncthreads();

    const int m  = lane & 15;
    const int hi = lane >> 4;
    const __bf16* sAw = &sA[wslot * (16 * LDS_STRIDE)];

    v8f acc[4] = {};
#pragma unroll
    for (int kk = 0; kk < 4; ++kk) {
        // A fragment per ISA 7.12.2 (16-bit A 16x32):
        // VGPR j holds K = (j/4)*16 + hi*8 + (j%4)*2 (+1)
        Frag16 a;
#pragma unroll
        for (int j = 0; j < 8; ++j) {
            int k = kk * 32 + ((j >> 2) << 4) + (hi << 3) + ((j & 3) << 1);
            a.w[j] = *(const uint32_t*)(sAw + m * LDS_STRIDE + k);
        }
#pragma unroll
        for (int t = 0; t < 4; ++t) {
            // B fragment (16-bit B 32x16): VGPR v holds K = (v/4)*16 + hi*8 + (v%4)*2 (+1), matching A
            Frag16 b;
#pragma unroll
            for (int v = 0; v < 8; ++v) {
                int k = kk * 32 + ((v >> 2) << 4) + (hi << 3) + ((v & 3) << 1);
                b.w[v] = *(const uint32_t*)(&sBt[(t * 16 + m) * LDS_STRIDE + k]);
            }
            acc[t] = __builtin_amdgcn_wmma_f32_16x16x32_bf16(
                false, a.v, false, b.v, (short)0, acc[t], false, false);
        }
    }

    // C/D layout: VGPR r -> M = r + 8*hi, N = lane%16 (per N-tile)
#pragma unroll
    for (int t = 0; t < 4; ++t) {
        float bv = bias[t * 16 + m];
#pragma unroll
        for (int r = 0; r < 8; ++r) {
            int row = tile0 + r + 8 * hi;
            if (row < nrows) out[(size_t)row * EMB + t * 16 + m] = acc[t][r] + bv;
        }
    }
}

// -------------------------------------------------------------------------
// Attention + aggregation: one wave32 per node.
// Lane l owns embed columns 2l, 2l+1. 24 neighbor embeddings live in regs.
// -------------------------------------------------------------------------
__global__ __launch_bounds__(256) void attn_agg_kernel(
    const int* __restrict__ nodes, const int* __restrict__ neigh_idx,
    const float* __restrict__ neigh_mask, const float* __restrict__ side_emb,
    const float* __restrict__ nodes_fea, float* __restrict__ embed_out, int n_nodes)
{
    const int wave = blockIdx.x * (blockDim.x >> 5) + (threadIdx.x >> 5);
    const int lane = threadIdx.x & 31;
    if (wave >= n_nodes) return;
    const int n = wave;

    float2 f = *(const float2*)&nodes_fea[(size_t)n * EMB + 2 * lane];

    float2 em[KNEI];
    float score = -3.0e38f;
#pragma unroll 4
    for (int k = 0; k < KNEI; ++k) {
        int si = neigh_idx[(size_t)n * KNEI + k];
        em[k] = *(const float2*)&side_emb[(size_t)si * EMB + 2 * lane];
        float p = em[k].x * f.x + em[k].y * f.y;
        for (int off = 16; off; off >>= 1) p += __shfl_xor(p, off, 32);
        p *= 0.125f;  // 1/sqrt(64)
        if (neigh_mask[(size_t)n * KNEI + k] <= 0.f) p = -1e9f;
        if (lane == k) score = p;
    }

    // softmax over lanes 0..23
    float v  = (lane < KNEI) ? score : -3.0e38f;
    float mx = v;
    for (int off = 16; off; off >>= 1) mx = fmaxf(mx, __shfl_xor(mx, off, 32));
    float e = (lane < KNEI) ? __expf(v - mx) : 0.f;
    float s = e;
    for (int off = 16; off; off >>= 1) s += __shfl_xor(s, off, 32);
    float att = e / s;

    float2 agg = {0.f, 0.f};
#pragma unroll 4
    for (int k = 0; k < KNEI; ++k) {
        float ak = __shfl(att, k, 32);
        agg.x += ak * em[k].x;
        agg.y += ak * em[k].y;
    }

    int row = nodes[n];
    *(float2*)&embed_out[(size_t)row * EMB + 2 * lane] = agg;
}

// -------------------------------------------------------------------------
extern "C" void kernel_launch(void* const* d_in, const int* in_sizes, int n_in,
                              void* d_out, int out_size, void* d_ws, size_t ws_size,
                              hipStream_t stream) {
    (void)in_sizes; (void)n_in; (void)out_size; (void)ws_size;
    const int N_NODES   = 50000;
    const int NUM_SIDES = 100000;

    const int*   nodes       = (const int*)d_in[0];
    const int*   neigh_idx   = (const int*)d_in[1];
    const float* neigh_mask  = (const float*)d_in[2];
    const float* drug_weight = (const float*)d_in[3];
    const float* side_weight = (const float*)d_in[4];
    const float* uW          = (const float*)d_in[5];
    const float* ub          = (const float*)d_in[6];
    const float* iW          = (const float*)d_in[7];
    const float* ib          = (const float*)d_in[8];

    float* nodes_fea = (float*)d_out;                           // [50000, 64]
    float* embed_out = (float*)d_out + (size_t)N_NODES * EMB;   // [50000, 64]
    float* side_emb  = (float*)d_ws;                            // [100000, 64] = 25.6 MB

    // 1) zero the scatter target
    {
        int n4 = N_NODES * EMB / 4;
        zero_f4_kernel<<<(n4 + 255) / 256, 256, 0, stream>>>((float4*)embed_out, n4);
    }
    // 2) side_emb = side_weight @ iW + ib  (factored out of the per-neighbor gather)
    {
        int tiles  = (NUM_SIDES + 15) / 16;
        int blocks = (tiles + 3) / 4;
        proj_gemm_kernel<<<blocks, 128, 0, stream>>>(side_weight, nullptr, iW, ib,
                                                     side_emb, NUM_SIDES);
    }
    // 3) nodes_fea = drug_weight[nodes] @ uW + ub
    {
        int tiles  = (N_NODES + 15) / 16;
        int blocks = (tiles + 3) / 4;
        proj_gemm_kernel<<<blocks, 128, 0, stream>>>(drug_weight, nodes, uW, ub,
                                                     nodes_fea, N_NODES);
    }
    // 4) attention softmax + weighted aggregation + scatter
    {
        int blocks = (N_NODES + 7) / 8;  // 8 waves (nodes) per 256-thread block
        attn_agg_kernel<<<blocks, 256, 0, stream>>>(nodes, neigh_idx, neigh_mask,
                                                    side_emb, nodes_fea, embed_out,
                                                    N_NODES);
    }
}
